// MultiLayerGAT_59983513256397
// MI455X (gfx1250) — matrix operand order repaired and checked by reading, not compiled, not run on hardware
//
#include <hip/hip_runtime.h>
#include <hip/hip_bf16.h>
#include <math.h>

// ---- Problem constants (from reference) ----
#define NN      50000
#define EE      800000
#define E2      (EE + NN)        // edges + self loops = 850000
#define IN_F    128
#define HID     64
#define HEADS   4
#define FOUT    (HEADS * HID)    // 256
#define NEG_SLOPE 0.2f

typedef __attribute__((ext_vector_type(16))) _Float16 v16h;
typedef __attribute__((ext_vector_type(8)))  _Float16 v8h;
typedef __attribute__((ext_vector_type(8)))  float    v8f;

// ---- order-preserving float <-> uint mapping for atomicMax over floats ----
__device__ __forceinline__ unsigned f2ord(float f) {
    unsigned u = __float_as_uint(f);
    return (u & 0x80000000u) ? ~u : (u | 0x80000000u);
}
__device__ __forceinline__ float ord2f(unsigned o) {
    unsigned u = (o & 0x80000000u) ? (o & 0x7fffffffu) : ~o;
    return __uint_as_float(u);
}

// ---------------- elementwise helpers ----------------
__global__ void k_cvt_f32_to_f16(const float* __restrict__ src,
                                 _Float16* __restrict__ dst, long long n) {
    long long i = (long long)blockIdx.x * blockDim.x + threadIdx.x;
    if (i < n) dst[i] = (_Float16)src[i];
}

__global__ void k_zero_f32(float* __restrict__ p, long long n) {
    long long i = (long long)blockIdx.x * blockDim.x + threadIdx.x;
    if (i < n) p[i] = 0.0f;
}

__global__ void k_init_segments(unsigned* __restrict__ mord,
                                float* __restrict__ denom, long long n) {
    long long i = (long long)blockIdx.x * blockDim.x + threadIdx.x;
    if (i < n) { mord[i] = f2ord(-INFINITY); denom[i] = 0.0f; }
}

// append self-loops, narrow int64 -> int32
__global__ void k_build_edges(const long long* __restrict__ e64,
                              int* __restrict__ srcI, int* __restrict__ dstI) {
    long long i = (long long)blockIdx.x * blockDim.x + threadIdx.x;
    if (i >= E2) return;
    if (i < EE) { srcI[i] = (int)e64[i]; dstI[i] = (int)e64[EE + i]; }
    else        { int n = (int)(i - EE); srcI[i] = n; dstI[i] = n; }
}

// ---------------- WMMA GEMM: C[M,256] = A[M,K] @ B[K,256], f16 in, f32 out ----
// One wave per 16x16 output tile; 8 waves per block.
__global__ __launch_bounds__(256)
void k_gemm_wmma_f16(const _Float16* __restrict__ A,
                     const _Float16* __restrict__ B,
                     float* __restrict__ C, int M, int K) {
    const int lane = threadIdx.x & 31;
    const long long gw = (long long)blockIdx.x * 8 + (threadIdx.x >> 5);
    const int mt = (int)(gw >> 4);          // 16 col-tiles across FOUT=256
    const int nt = (int)(gw & 15);
    const int m0 = mt * 16, n0 = nt * 16;
    if (m0 >= M) return;

    const int r  = lane & 15;               // A row within tile
    const int kb = (lane < 16) ? 0 : 8;     // K-group base per ISA A layout
    const _Float16* __restrict__ arow = A + (size_t)(m0 + r) * K;

    v8f acc = {};
    for (int k0 = 0; k0 < K; k0 += 32) {
        // A frag: row r, halves = {k0+kb .. +7} and {k0+kb+16 .. +23}
        v8h alo = *(const v8h*)(arow + k0 + kb);
        v8h ahi = *(const v8h*)(arow + k0 + kb + 16);
        // B frag: col n0+r, halves = {k0+kb .. +7} and {k0+kb+16 .. +23}
        const _Float16* __restrict__ brow = B + (size_t)(k0 + kb) * FOUT + n0 + r;
        v16h af, bf;
#pragma unroll
        for (int i = 0; i < 8; ++i) {
            af[i] = alo[i]; af[i + 8] = ahi[i];
            bf[i] = brow[(size_t)i * FOUT]; bf[i + 8] = brow[(size_t)(i + 16) * FOUT];
        }
        acc = __builtin_amdgcn_wmma_f32_16x16x32_f16(
            false, af, false, bf, (short)0, acc, false, false);
    }
    // D layout: acc[rr] -> row m0 + 8*(lane/16) + rr, col n0 + lane%16
    const int mrow = m0 + ((lane >> 4) << 3);
    const int ncol = n0 + (lane & 15);
#pragma unroll
    for (int rr = 0; rr < 8; ++rr)
        C[(size_t)(mrow + rr) * FOUT + ncol] = acc[rr];
}

// ---------------- per-node attention logits: alpha_{src,dst}[n,h] ----------
__global__ void k_alpha(const float* __restrict__ h,
                        const float* __restrict__ a_src,
                        const float* __restrict__ a_dst,
                        float* __restrict__ asrc, float* __restrict__ adst) {
    long long t = (long long)blockIdx.x * blockDim.x + threadIdx.x;
    if (t >= (long long)NN * HEADS) return;
    const int n = (int)(t >> 2), hh = (int)(t & 3);
    const float* __restrict__ hp = h + (size_t)n * FOUT + hh * HID;
    const float* __restrict__ as = a_src + hh * HID;
    const float* __restrict__ ad = a_dst + hh * HID;
    float s1 = 0.f, s2 = 0.f;
#pragma unroll 4
    for (int d = 0; d < HID; ++d) { float v = hp[d]; s1 += v * as[d]; s2 += v * ad[d]; }
    asrc[t] = s1; adst[t] = s2;
}

// ---------------- edge pass 1: segment max (ordered-uint atomicMax) --------
__global__ void k_edge_max(const int* __restrict__ srcI, const int* __restrict__ dstI,
                           const float* __restrict__ asrc, const float* __restrict__ adst,
                           unsigned* __restrict__ mord) {
    long long t = (long long)blockIdx.x * blockDim.x + threadIdx.x;
    if (t >= (long long)E2 * HEADS) return;
    const long long e = t >> 2; const int hh = (int)(t & 3);
    const int s = srcI[e], d = dstI[e];
    float ee = asrc[(size_t)s * HEADS + hh] + adst[(size_t)d * HEADS + hh];
    ee = (ee > 0.f) ? ee : NEG_SLOPE * ee;                // leaky_relu
    atomicMax(mord + (size_t)d * HEADS + hh, f2ord(ee));
}

// ---------------- edge pass 2: exp(e - m), segment sum --------------------
__global__ void k_edge_expsum(const int* __restrict__ srcI, const int* __restrict__ dstI,
                              const float* __restrict__ asrc, const float* __restrict__ adst,
                              const unsigned* __restrict__ mord,
                              float* __restrict__ exw, float* __restrict__ denom) {
    long long t = (long long)blockIdx.x * blockDim.x + threadIdx.x;
    if (t >= (long long)E2 * HEADS) return;
    const long long e = t >> 2; const int hh = (int)(t & 3);
    const int s = srcI[e], d = dstI[e];
    float ee = asrc[(size_t)s * HEADS + hh] + adst[(size_t)d * HEADS + hh];
    ee = (ee > 0.f) ? ee : NEG_SLOPE * ee;
    const float m = ord2f(mord[(size_t)d * HEADS + hh]);
    const float ex = __expf(ee - m);
    exw[t] = ex;
    atomicAdd(denom + (size_t)d * HEADS + hh, ex);
}

// ---------------- edge pass 3: weighted scatter-add aggregation -----------
// thread per (edge, channel): coalesced row gather of h[src], coalesced
// atomic row update of agg[dst] — resolves in L2 (h = 51 MB << 192 MB L2).
__global__ void k_edge_agg(const int* __restrict__ srcI, const int* __restrict__ dstI,
                           const float* __restrict__ exw, const float* __restrict__ denom,
                           const float* __restrict__ h, float* __restrict__ agg) {
    long long t = (long long)blockIdx.x * blockDim.x + threadIdx.x;
    if (t >= (long long)E2 * FOUT) return;
    const long long e = t >> 8; const int c = (int)(t & 255); const int hh = c >> 6;
    const int s = srcI[e], d = dstI[e];
    const float alpha = exw[(size_t)e * HEADS + hh] /
                        (denom[(size_t)d * HEADS + hh] + 1e-16f);
    atomicAdd(agg + (size_t)d * FOUT + c, h[(size_t)s * FOUT + c] * alpha);
}

// ---------------- finalize: +bias, relu; layer1 emits f16 for next GEMM ---
__global__ void k_finalize_to_f16(const float* __restrict__ agg,
                                  const float* __restrict__ b,
                                  _Float16* __restrict__ outh) {
    long long i = (long long)blockIdx.x * blockDim.x + threadIdx.x;
    if (i >= (long long)NN * FOUT) return;
    float v = agg[i] + b[(int)(i & 255)];
    outh[i] = (_Float16)(v > 0.f ? v : 0.f);
}

__global__ void k_finalize_inplace(float* __restrict__ out, const float* __restrict__ b) {
    long long i = (long long)blockIdx.x * blockDim.x + threadIdx.x;
    if (i >= (long long)NN * FOUT) return;
    float v = out[i] + b[(int)(i & 255)];
    out[i] = (v > 0.f) ? v : 0.f;
}

// ===========================================================================
extern "C" void kernel_launch(void* const* d_in, const int* in_sizes, int n_in,
                              void* d_out, int out_size, void* d_ws, size_t ws_size,
                              hipStream_t stream) {
    const float*     x      = (const float*)d_in[0];
    const long long* eidx   = (const long long*)d_in[1];
    const float*     W1     = (const float*)d_in[2];
    const float*     a_src1 = (const float*)d_in[3];
    const float*     a_dst1 = (const float*)d_in[4];
    const float*     b1     = (const float*)d_in[5];
    const float*     W2     = (const float*)d_in[6];
    const float*     a_src2 = (const float*)d_in[7];
    const float*     a_dst2 = (const float*)d_in[8];
    const float*     b2     = (const float*)d_in[9];
    float* out = (float*)d_out;

    // ---- workspace carving (256B aligned) ----
    char* ws = (char*)d_ws; size_t off = 0;
    auto alloc = [&](size_t bytes) -> char* {
        char* p = ws + off; off += (bytes + 255) & ~(size_t)255; return p;
    };
    _Float16* xh    = (_Float16*)alloc((size_t)NN * IN_F * 2);
    _Float16* W1h   = (_Float16*)alloc((size_t)IN_F * FOUT * 2);
    _Float16* W2h   = (_Float16*)alloc((size_t)FOUT * FOUT * 2);
    float*    h     = (float*)   alloc((size_t)NN * FOUT * 4);   // projection (both layers)
    _Float16* h1h   = (_Float16*)alloc((size_t)NN * FOUT * 2);   // layer-2 GEMM input
    float*    asrc  = (float*)   alloc((size_t)NN * HEADS * 4);
    float*    adst  = (float*)   alloc((size_t)NN * HEADS * 4);
    unsigned* mord  = (unsigned*)alloc((size_t)NN * HEADS * 4);
    float*    denom = (float*)   alloc((size_t)NN * HEADS * 4);
    float*    exw   = (float*)   alloc((size_t)E2 * HEADS * 4);
    float*    agg   = (float*)   alloc((size_t)NN * FOUT * 4);   // layer-1 accumulator
    int*      srcI  = (int*)     alloc((size_t)E2 * 4);
    int*      dstI  = (int*)     alloc((size_t)E2 * 4);
    (void)ws_size; (void)in_sizes; (void)n_in; (void)out_size;

    const int T = 256;
    auto blocks = [](long long n, int t) { return (unsigned)((n + t - 1) / t); };

    const long long nNF  = (long long)NN * FOUT;     // 12.8M
    const long long nNH  = (long long)NN * HEADS;    // 200K
    const long long nEH  = (long long)E2 * HEADS;    // 3.4M
    const long long nEC  = (long long)E2 * FOUT;     // 217.6M

    // ---- one-time conversions / edge list ----
    k_cvt_f32_to_f16<<<blocks((long long)NN * IN_F, T), T, 0, stream>>>(x, xh, (long long)NN * IN_F);
    k_cvt_f32_to_f16<<<blocks((long long)IN_F * FOUT, T), T, 0, stream>>>(W1, W1h, (long long)IN_F * FOUT);
    k_cvt_f32_to_f16<<<blocks((long long)FOUT * FOUT, T), T, 0, stream>>>(W2, W2h, (long long)FOUT * FOUT);
    k_build_edges<<<blocks(E2, T), T, 0, stream>>>(eidx, srcI, dstI);

    // ================= layer 1 =================
    k_gemm_wmma_f16<<<blocks((long long)(NN / 16) * 16, 8) /* waves/8 */, T, 0, stream>>>(xh, W1h, h, NN, IN_F);
    k_alpha<<<blocks(nNH, T), T, 0, stream>>>(h, a_src1, a_dst1, asrc, adst);
    k_init_segments<<<blocks(nNH, T), T, 0, stream>>>(mord, denom, nNH);
    k_edge_max<<<blocks(nEH, T), T, 0, stream>>>(srcI, dstI, asrc, adst, mord);
    k_edge_expsum<<<blocks(nEH, T), T, 0, stream>>>(srcI, dstI, asrc, adst, mord, exw, denom);
    k_zero_f32<<<blocks(nNF, T), T, 0, stream>>>(agg, nNF);
    k_edge_agg<<<blocks(nEC, T), T, 0, stream>>>(srcI, dstI, exw, denom, h, agg);
    k_finalize_to_f16<<<blocks(nNF, T), T, 0, stream>>>(agg, b1, h1h);

    // ================= layer 2 =================
    k_gemm_wmma_f16<<<blocks((long long)(NN / 16) * 16, 8), T, 0, stream>>>(h1h, W2h, h, NN, FOUT);
    k_alpha<<<blocks(nNH, T), T, 0, stream>>>(h, a_src2, a_dst2, asrc, adst);
    k_init_segments<<<blocks(nNH, T), T, 0, stream>>>(mord, denom, nNH);
    k_edge_max<<<blocks(nEH, T), T, 0, stream>>>(srcI, dstI, asrc, adst, mord);
    k_edge_expsum<<<blocks(nEH, T), T, 0, stream>>>(srcI, dstI, asrc, adst, mord, exw, denom);
    k_zero_f32<<<blocks(nNF, T), T, 0, stream>>>(out, nNF);        // aggregate straight into d_out
    k_edge_agg<<<blocks(nEC, T), T, 0, stream>>>(srcI, dstI, exw, denom, h, out);
    k_finalize_inplace<<<blocks(nNF, T), T, 0, stream>>>(out, b2);
}
